// drosophRNN_15436112462539
// MI455X (gfx1250) — matrix operand order repaired and checked by reading, not compiled, not run on hardware
//
#include <hip/hip_runtime.h>

// ---------------------------------------------------------------------------
// CDNA5 (gfx1250) ring-attractor RNN scan.
//   - 4 blocks x 256 threads (8 waves, wave32). Block b owns batches [16b,16b+16).
//   - All einsums -> v_wmma_f32_16x16x32_f16 (f16 in, f32 accumulate).
//   - Weights pre-packed once per launch into f16 WMMA A-fragment layout in d_ws,
//     then HOISTED INTO REGISTERS (24 tiles = 192 VGPRs per wave) before the
//     5000-step scan loop: each phase's critical path is ds_load -> wmma only.
//   - State kept in LDS (f32), converted to f16 B-fragments in LDS each step.
//   - Step outputs streamed with non-temporal stores (384 MB never re-read).
// ---------------------------------------------------------------------------

typedef __attribute__((ext_vector_type(16))) _Float16 v16h;
typedef __attribute__((ext_vector_type(8)))  float    v8f;

#define S_LEN 5000
#define B_TOT 64
#define NB    16          // batches per block (WMMA N dim)
#define NCHP  384         // padded channels: HD[0,128) AVp[128,192) AVm[192,256) D7[256,384)
#define NTILES 168        // G0:7x8=56, G1:7x4=28, G2:7x4=28, G3:7x8=56
// tile bases in d_ws
#define G0_BASE 0
#define G1_BASE 56
#define G2_BASE 84
#define G3_BASE 112

// ---------------------------------------------------------------------------
// Pack all weight matrices (pre-scaled, DT folded in) into f16 WMMA A-fragment
// layout:  ws[(tile*32 + lane)*16 + e],  lane = m_local + 16*((k>>3)&1),
//          e = ((k>>4)<<3) | (k&7)   (ISA 7.12.2, 16-bit A 16x32).
// ---------------------------------------------------------------------------
__global__ void rnn_prep_weights(const float* __restrict__ W_HD_HD,
                                 const float* __restrict__ W_HD_AVp,
                                 const float* __restrict__ W_HD_AVm,
                                 const float* __restrict__ W_AVp_HD,
                                 const float* __restrict__ W_AVm_HD,
                                 const float* __restrict__ W_D7_HD,
                                 const float* __restrict__ W_D7_D7,
                                 const float* __restrict__ W_HD_D7,
                                 _Float16* __restrict__ ws)
{
    int g = blockIdx.x * blockDim.x + threadIdx.x;
    if (g >= NTILES * 512) return;
    int tile = g >> 9;
    int rem  = g & 511;
    int lane = rem >> 4;
    int e    = rem & 15;
    int ml   = lane & 15;                       // m within 16-row tile
    int hi   = lane >> 4;
    int kl   = ((e & 8) << 1) | (hi << 3) | (e & 7);   // k within 32-wide tile

    int gemm, mt, kt;
    if      (tile < G1_BASE) { gemm = 0; mt = tile >> 3;             kt = tile & 7; }
    else if (tile < G2_BASE) { gemm = 1; mt = (tile - G1_BASE) >> 2; kt = (tile - G1_BASE) & 3; }
    else if (tile < G3_BASE) { gemm = 2; mt = (tile - G2_BASE) >> 2; kt = (tile - G2_BASE) & 3; }
    else                     { gemm = 3; mt = (tile - G3_BASE) >> 3; kt = (tile - G3_BASE) & 7; }
    int m = mt * 16 + ml;
    int k = kt * 32 + kl;

    const float DT        = 0.001f;
    const float W_HD_HD_S = 10.5f;                    // ALPHA_TILDE + 1/(kp+ky)
    const float W_HD_AV_S = 0.70710678118654752f;     // sqrt(2)*ky/(ky+kp)
    const float W_HD_D7_S = 0.5f;                     // 1/(kp+ky)
    const float DT_TAU_AV = 0.1f;                     // DT/TAU_AV
    const float DT_TAU_D7 = 1.0f;                     // DT/TAU_D7

    float v = 0.f;
    if (gemm == 0) {            // phase1 main: [HD | AVp | AVm] -> HD  (x DT)
        if (m < 100) {
            if (k < 128)      { int j = k;       if (j < 100) v = DT * W_HD_HD_S * W_HD_HD[m*100 + j]; }
            else if (k < 192) { int j = k - 128; if (j < 50)  v = DT * W_HD_AV_S * W_HD_AVp[m*50 + j]; }
            else              { int j = k - 192; if (j < 50)  v = DT * W_HD_AV_S * W_HD_AVm[m*50 + j]; }
        }
    } else if (gemm == 1) {     // phase1 multiplicative: relu(D7) -> HD  (x DT*W_HD_D7_S)
        if (m < 100 && k < 100) v = DT * W_HD_D7_S * W_HD_D7[m*100 + k];
    } else if (gemm == 2) {     // phase2 AV (stacked +/-): HD_new -> AV  (x DT/TAU_AV)
        if (k < 100) {
            if      (m < 50)  v = DT_TAU_AV * W_AVp_HD[m*100 + k];
            else if (m < 100) v = DT_TAU_AV * W_AVm_HD[(m-50)*100 + k];
        }
    } else {                    // phase2 D7: [relu(HD_new) | relu(D7)] -> D7  (x DT/TAU_D7)
        if (m < 100) {
            if (k < 128)      { int j = k;       if (j < 100) v = DT_TAU_D7 * W_D7_HD[m*100 + j]; }
            else              { int j = k - 128; if (j < 100) v = DT_TAU_D7 * W_D7_D7[m*100 + j]; }
        }
    }
    ws[(tile * 32 + lane) * 16 + e] = (_Float16)v;
}

// ---------------------------------------------------------------------------
// Main scan kernel. One block = 16 batches, 8 waves; waves 0..6 each own one
// 16-row M-tile of every GEMM; wave 7 only helps with staging / stores.
// ---------------------------------------------------------------------------
__global__ void __launch_bounds__(256) rnn_scan(const float* __restrict__ xin,
                                                const float* __restrict__ r0,
                                                const _Float16* __restrict__ wsA,
                                                float* __restrict__ out)
{
    __shared__ float stF[NCHP * NB];                       // f32 state [chan][batch]
    __shared__ __align__(32) _Float16 stH[20 * 32 * 16];   // staged f16 B frags (20 tiles)
    __shared__ float gP[NB], gM[NB];                       // angular-velocity gates

    const int tid  = threadIdx.x;
    const int wave = tid >> 5;
    const int lane = tid & 31;
    const int n    = lane & 15;          // batch column within tile
    const int hi   = lane >> 4;
    const int blk  = blockIdx.x;         // 0..3

    // zero padded state, then load r0 (channel map: 300 -> padded 384)
    for (int i = tid; i < NCHP * NB; i += 256) stF[i] = 0.f;
    __syncthreads();
    for (int idx = tid; idx < 300 * NB; idx += 256) {
        int b = idx / 300, c = idx - b * 300;
        int pc = (c < 100) ? c : (c < 150) ? 128 + (c - 100)
               : (c < 200) ? 192 + (c - 150) : 256 + (c - 200);
        stF[pc * NB + b] = r0[(blk * NB + b) * 300 + c];
    }
    __syncthreads();

    // ---- hoist this wave's A-fragments (constant weights) into registers:
    //      24 tiles x 8 VGPRs = 192 VGPRs; only 2 waves/SIMD so pressure is fine.
    v8f zf = {};
    v16h A0[8], A1[4], A2[4], A3[8];
#pragma unroll
    for (int kt = 0; kt < 8; ++kt) A0[kt] = (v16h)zf;
#pragma unroll
    for (int kt = 0; kt < 4; ++kt) { A1[kt] = (v16h)zf; A2[kt] = (v16h)zf; }
#pragma unroll
    for (int kt = 0; kt < 8; ++kt) A3[kt] = (v16h)zf;
    if (wave < 7) {
#pragma unroll
        for (int kt = 0; kt < 8; ++kt)
            A0[kt] = *(const v16h*)&wsA[((G0_BASE + wave * 8 + kt) * 32 + lane) * 16];
#pragma unroll
        for (int kt = 0; kt < 4; ++kt)
            A1[kt] = *(const v16h*)&wsA[((G1_BASE + wave * 4 + kt) * 32 + lane) * 16];
#pragma unroll
        for (int kt = 0; kt < 4; ++kt)
            A2[kt] = *(const v16h*)&wsA[((G2_BASE + wave * 4 + kt) * 32 + lane) * 16];
#pragma unroll
        for (int kt = 0; kt < 8; ++kt)
            A3[kt] = *(const v16h*)&wsA[((G3_BASE + wave * 8 + kt) * 32 + lane) * 16];
    }

    const float C_HD = 1.f - 10.25f * 0.001f;   // 1 - ALPHA*DT
    const float C_AV = 0.9f;                    // 1 - DT/TAU_AV
    const float C_D7 = 0.f;                     // 1 - DT/TAU_D7

    for (int t = 0; t < S_LEN; ++t) {
        // ---- stage phase-1 B fragments ----------------------------------
        // slots 0..7 : chans 0..255 raw   ([r_HD | r_AVp | r_AVm])
        // slots 8..11: chans 256..383 relu (relu(r_D7))
        if (tid < NB) {
            float av = xin[((blk * NB + tid) * S_LEN + t) * 2 + 1];
            gP[tid] =  av * 1000.0f;    // av/DT + AV_OFFSET(=0)
            gM[tid] = -av * 1000.0f;
        }
        for (int idx = tid; idx < 12 * 512; idx += 256) {
            int slot = idx >> 9, rem = idx & 511;
            int l = rem >> 4, e = rem & 15;
            int kl = ((e & 8) << 1) | ((l >> 4) << 3) | (e & 7);   // B layout: table order, matches A
            float v = stF[(slot * 32 + kl) * NB + (l & 15)];
            if (slot >= 8) v = fmaxf(v, 0.f);
            stH[(slot * 32 + l) * 16 + e] = (_Float16)v;
        }
        __syncthreads();

        // ---- phase 1: r_HD_new ------------------------------------------
        if (wave < 7) {
            v8f acc1 = {}; v8f acc2 = {};
#pragma unroll
            for (int kt = 0; kt < 8; ++kt) {
                v16h b = *(const v16h*)&stH[(kt * 32 + lane) * 16];
                acc1 = __builtin_amdgcn_wmma_f32_16x16x32_f16(false, A0[kt], false, b,
                                                              (short)0, acc1, false, false);
            }
#pragma unroll
            for (int kt = 0; kt < 4; ++kt) {
                v16h b = *(const v16h*)&stH[((8 + kt) * 32 + lane) * 16];
                acc2 = __builtin_amdgcn_wmma_f32_16x16x32_f16(false, A1[kt], false, b,
                                                              (short)0, acc2, false, false);
            }
#pragma unroll
            for (int r = 0; r < 8; ++r) {        // C layout: m = r + 8*hi (+ tile)
                int m = wave * 16 + r + 8 * hi;
                float old = stF[m * NB + n];
                stF[m * NB + n] = old * C_HD + acc1[r] + acc2[r] * old;
            }
        }
        __syncthreads();

        // ---- stage phase-2 B fragments ----------------------------------
        // slots 12..15: HD_new raw ; slots 16..19: relu(HD_new)
        for (int idx = tid; idx < 8 * 512; idx += 256) {
            int slot = idx >> 9, rem = idx & 511;
            int l = rem >> 4, e = rem & 15;
            int kl = ((e & 8) << 1) | ((l >> 4) << 3) | (e & 7);
            float v = stF[((slot & 3) * 32 + kl) * NB + (l & 15)];
            if (slot >= 4) v = fmaxf(v, 0.f);
            stH[((12 + slot) * 32 + l) * 16 + e] = (_Float16)v;
        }
        __syncthreads();

        // ---- phase 2: AV and Delta7 updates -----------------------------
        if (wave < 7) {
            v8f accA = {}; v8f accD = {};
#pragma unroll
            for (int kt = 0; kt < 4; ++kt) {
                v16h b = *(const v16h*)&stH[((12 + kt) * 32 + lane) * 16];
                accA = __builtin_amdgcn_wmma_f32_16x16x32_f16(false, A2[kt], false, b,
                                                              (short)0, accA, false, false);
            }
#pragma unroll
            for (int kt = 0; kt < 8; ++kt) {
                v16h b = (kt < 4)
                       ? *(const v16h*)&stH[((16 + kt) * 32 + lane) * 16]        // relu(HD_new)
                       : *(const v16h*)&stH[((8 + (kt - 4)) * 32 + lane) * 16];  // relu(D7_old)
                accD = __builtin_amdgcn_wmma_f32_16x16x32_f16(false, A3[kt], false, b,
                                                              (short)0, accD, false, false);
            }
#pragma unroll
            for (int r = 0; r < 8; ++r) {
                int m = wave * 16 + r + 8 * hi;
                if (m < 50) {                                  // AV+ rows
                    float old = stF[(128 + m) * NB + n];
                    stF[(128 + m) * NB + n] = old * C_AV + gP[n] * accA[r];
                } else if (m < 100) {                          // AV- rows
                    float old = stF[(192 + m - 50) * NB + n];
                    stF[(192 + m - 50) * NB + n] = old * C_AV + gM[n] * accA[r];
                }
                float oldd = stF[(256 + m) * NB + n];          // D7 rows (pad rows stay 0)
                stF[(256 + m) * NB + n] = oldd * C_D7 + accD[r];
            }
        }
        __syncthreads();

        // ---- write step output (coalesced, non-temporal: never re-read) --
        for (int idx = tid; idx < 300 * NB; idx += 256) {
            int b = idx / 300, c = idx - b * 300;
            int pc = (c < 100) ? c : (c < 150) ? 128 + (c - 100)
                   : (c < 200) ? 192 + (c - 150) : 256 + (c - 200);
            __builtin_nontemporal_store(stF[pc * NB + b],
                &out[(size_t)((blk * NB + b) * S_LEN + t) * 300 + c]);
        }
        // next iteration's staging barrier orders these reads vs phase-1 writes
    }

    // ---- final carry state r_full_curr ---------------------------------
    for (int idx = tid; idx < 300 * NB; idx += 256) {
        int b = idx / 300, c = idx - b * 300;
        int pc = (c < 100) ? c : (c < 150) ? 128 + (c - 100)
               : (c < 200) ? 192 + (c - 150) : 256 + (c - 200);
        out[(size_t)B_TOT * S_LEN * 300 + (size_t)(blk * NB + b) * 300 + c] = stF[pc * NB + b];
    }
}

// ---------------------------------------------------------------------------
extern "C" void kernel_launch(void* const* d_in, const int* in_sizes, int n_in,
                              void* d_out, int out_size, void* d_ws, size_t ws_size,
                              hipStream_t stream)
{
    (void)in_sizes; (void)n_in; (void)out_size; (void)ws_size;
    const float* xin      = (const float*)d_in[0];   // inputs  (64,5000,2)
    const float* r0       = (const float*)d_in[1];   // r0      (1,64,300)
    const float* W_HD_HD  = (const float*)d_in[2];
    const float* W_HD_AVp = (const float*)d_in[3];
    const float* W_HD_AVm = (const float*)d_in[4];
    const float* W_AVp_HD = (const float*)d_in[5];
    const float* W_AVm_HD = (const float*)d_in[6];
    const float* W_D7_HD  = (const float*)d_in[7];
    const float* W_D7_D7  = (const float*)d_in[8];
    const float* W_HD_D7  = (const float*)d_in[9];
    _Float16* wsA = (_Float16*)d_ws;                 // 168 KB packed f16 A-fragments
    float* out = (float*)d_out;

    rnn_prep_weights<<<(NTILES * 512 + 255) / 256, 256, 0, stream>>>(
        W_HD_HD, W_HD_AVp, W_HD_AVm, W_AVp_HD, W_AVm_HD, W_D7_HD, W_D7_D7, W_HD_D7, wsA);
    rnn_scan<<<4, 256, 0, stream>>>(xin, r0, wsA, out);
}
